// GeneratorCrossAttention_76201309765723
// MI455X (gfx1250) — matrix operand order repaired and checked by reading, not compiled, not run on hardware
//
#include <hip/hip_runtime.h>
#include <cmath>

typedef __bf16 bf16;
typedef __attribute__((ext_vector_type(16))) __bf16 v16bf;
typedef __attribute__((ext_vector_type(8)))  __bf16 v8bf;
typedef __attribute__((ext_vector_type(8)))  float  v8f;

#define BB 8
#define NQ 1024
#define NK 256
#define HH 8
#define TQ (BB*NQ)      // 8192 query tokens
#define TK (BB*NK)      // 2048 kv tokens
#define CI 64           // padded per-component input dim (48 -> 64)
#define DA 544          // augmented attention dim: 11*48=528 padded to 17*32
#define ALPHA 0.05103103630798288f  // 1/sqrt(8*C)=1/sqrt(384)

__device__ __constant__ int GRADE[16]   = {0,1,1,1,1,2,2,2,2,2,2,3,3,3,3,4};
__device__ __constant__ int SLOT_OF[16] = {0,-1,1,2,3,-1,-1,-1,4,5,6,8,9,10,7,-1};
__device__ __constant__ int SLOT_TO_J[11] = {0,2,3,4,8,9,10,14,11,12,13};

// ---- async global -> LDS staging (CDNA5 ASYNCcnt path), with safe fallback ----
typedef int v4i __attribute__((vector_size(16)));
typedef __attribute__((address_space(1))) v4i as1_v4i;
typedef __attribute__((address_space(3))) v4i as3_v4i;
#if __has_builtin(__builtin_amdgcn_global_load_async_to_lds_b128) && \
    __has_builtin(__builtin_amdgcn_s_wait_asynccnt)
#define USE_ASYNC_LDS 1
#endif

__device__ __forceinline__ void async_copy16(bf16* lds_dst, const bf16* gsrc) {
#ifdef USE_ASYNC_LDS
  __builtin_amdgcn_global_load_async_to_lds_b128((as1_v4i*)gsrc, (as3_v4i*)lds_dst, 0, 0);
#else
  *(v16bf*)lds_dst = *(const v16bf*)gsrc;
#endif
}
// threshold waits: allow the newest N async ops to remain in flight (double buffering)
__device__ __forceinline__ void async_wait0()  {
#ifdef USE_ASYNC_LDS
  __builtin_amdgcn_s_wait_asynccnt(0);
#endif
}
__device__ __forceinline__ void async_wait8()  {
#ifdef USE_ASYNC_LDS
  __builtin_amdgcn_s_wait_asynccnt(8);
#endif
}
__device__ __forceinline__ void async_wait16() {
#ifdef USE_ASYNC_LDS
  __builtin_amdgcn_s_wait_asynccnt(16);
#endif
}

// ---- WMMA helpers (CDNA5 16x16x32 bf16 -> f32) ----
__device__ __forceinline__ v8f wmma_bf16(v16bf a, v16bf b, v8f c) {
  return __builtin_amdgcn_wmma_f32_16x16x32_bf16(false, a, false, b, (short)0, c, false, false);
}

// A fragment (16x32, 16-bit): lane L: M=L&15; K = 16*half + {0..15} contiguous
__device__ __forceinline__ v16bf load_a(const bf16* base, int ld, int row, int k0, int half) {
  return *(const v16bf*)(base + (size_t)row * ld + k0 + 16 * half);
}
// B fragment (32x16, 16-bit): lane L: N=L&15; K = 16*half + {0..15} contiguous
__device__ __forceinline__ v16bf load_b(const bf16* base, int ld, int nrow, int k0, int half) {
  return *(const v16bf*)(base + (size_t)nrow * ld + k0 + 16 * half);
}

#define V8F_ZERO {0.f,0.f,0.f,0.f,0.f,0.f,0.f,0.f}

// ---------- precompute softplus(daa_scale) (tiny; keeps log1p out of hot kernels) ----------
__global__ void k_sp(const float* __restrict__ daa, float* __restrict__ sp) {
  int h = threadIdx.x;
  if (h < HH) {
    float x = daa[h];
    sp[h] = (x > 20.f) ? x : log1pf(__expf(x));
  }
}

// ---------- pack weights to bf16 (padded K) ----------
__global__ void k_packw(const float* __restrict__ Wq, const float* __restrict__ Wkv,
                        const float* __restrict__ Wo,
                        bf16* __restrict__ wq, bf16* __restrict__ wkv, bf16* __restrict__ wo) {
  int idx = blockIdx.x * 256 + threadIdx.x;
  if (idx < 5 * 384 * CI) {
    int i = idx & 63, row = idx >> 6;
    wq[idx] = (bf16)((i < 48) ? Wq[row * 48 + i] : 0.f);
    return;
  }
  idx -= 5 * 384 * CI;
  if (idx < 5 * 768 * CI) {
    int i = idx & 63, row = idx >> 6;
    wkv[idx] = (bf16)((i < 48) ? Wkv[row * 48 + i] : 0.f);
    return;
  }
  idx -= 5 * 768 * CI;
  if (idx < 5 * 48 * 384) wo[idx] = (bf16)Wo[idx];
}

// ---------- equi RMS norm + transpose-pack hidden to [t][j][64] bf16 ----------
__global__ __launch_bounds__(256) void k_norm(const float* __restrict__ hidden,
                                              const float* __restrict__ lnw,
                                              bf16* __restrict__ normed) {
  __shared__ float red[256];
  int t = blockIdx.x;
  const float* xp = hidden + (size_t)t * 768;
  float s = 0.f;
  for (int i = threadIdx.x; i < 768; i += 256) { float v = xp[i]; s += v * v; }
  red[threadIdx.x] = s;
  __syncthreads();
  for (int off = 128; off > 0; off >>= 1) {
    if (threadIdx.x < off) red[threadIdx.x] += red[threadIdx.x + off];
    __syncthreads();
  }
  float inv = rsqrtf(red[0] / 48.f + 1e-6f);
  bf16* op = normed + (size_t)t * (16 * CI);
  for (int idx = threadIdx.x; idx < 16 * CI; idx += 256) {
    int j = idx >> 6, i = idx & 63;
    float v = (i < 48) ? xp[i * 16 + j] * inv * lnw[i] : 0.f;
    op[idx] = (bf16)v;
  }
}

// ---------- transpose-pack vision to [t][j][64] bf16 ----------
__global__ void k_packv(const float* __restrict__ vis, bf16* __restrict__ visbf) {
  int idx = blockIdx.x * 256 + threadIdx.x;  // total TK*16*64, exact grid
  int i = idx & 63, j = (idx >> 6) & 15, t = idx >> 10;
  float v = (i < 48) ? vis[(size_t)t * 768 + i * 16 + j] : 0.f;
  visbf[idx] = (bf16)v;
}

// ---------- Q projection -> scaled augmented Q [bh][q][544] ----------
__global__ __launch_bounds__(32) void k_projq(const bf16* __restrict__ normed,
                                              const bf16* __restrict__ wq,
                                              const float* __restrict__ bq,
                                              const float* __restrict__ sp,
                                              bf16* __restrict__ Qaug) {
  int t0 = blockIdx.x * 16, o00 = blockIdx.y * 64, slot = blockIdx.z;
  int j = SLOT_TO_J[slot], g = GRADE[j];
  int lane = threadIdx.x & 31, half = lane >> 4, nI = lane & 15;
  const bf16* A = normed + j * CI;               // row stride 16*CI
  const bf16* Bm = wq + (size_t)g * 384 * CI;    // row stride CI
  v8f accs[4] = {V8F_ZERO, V8F_ZERO, V8F_ZERO, V8F_ZERO};
#pragma unroll
  for (int ks = 0; ks < 2; ks++) {
    v16bf a = load_a(A, 16 * CI, t0 + nI, ks * 32, half);
#pragma unroll
    for (int u = 0; u < 4; u++) {
      v16bf b = load_b(Bm, CI, o00 + u * 16 + nI, ks * 32, half);
      accs[u] = wmma_bf16(a, b, accs[u]);
    }
  }
#pragma unroll
  for (int u = 0; u < 4; u++) {
    int o0 = o00 + u * 16;
    int h = o0 / 48, cbase = o0 % 48;        // block-uniform (tiles never cross h)
    int o = o0 + nI, c = cbase + nI;
    float scale = (slot < 8) ? ALPHA : sp[h] * (2.f / 48.f);
    float bias = (j == 0) ? bq[o] : 0.f;
#pragma unroll
    for (int r = 0; r < 8; r++) {
      int t = t0 + r + 8 * half;
      int bI = t >> 10, q = t & 1023;
      Qaug[(((size_t)(bI * HH + h)) * NQ + q) * DA + slot * 48 + c] =
          (bf16)((accs[u][r] + bias) * scale);
    }
  }
}

// ---------- KV projection -> augmented K [bh][k][544] and V^T [bh][768][256] ----------
__global__ __launch_bounds__(32) void k_projkv(const bf16* __restrict__ visbf,
                                               const bf16* __restrict__ wkv,
                                               const float* __restrict__ bkv,
                                               bf16* __restrict__ Kaug,
                                               bf16* __restrict__ Vt) {
  int t0 = blockIdx.x * 16, o00 = blockIdx.y * 64, j = blockIdx.z;
  int g = GRADE[j];
  int lane = threadIdx.x & 31, half = lane >> 4, nI = lane & 15;
  const bf16* A = visbf + j * CI;
  const bf16* Bm = wkv + (size_t)g * 768 * CI;
  v8f accs[4] = {V8F_ZERO, V8F_ZERO, V8F_ZERO, V8F_ZERO};
#pragma unroll
  for (int ks = 0; ks < 2; ks++) {
    v16bf a = load_a(A, 16 * CI, t0 + nI, ks * 32, half);
#pragma unroll
    for (int u = 0; u < 4; u++) {
      v16bf b = load_b(Bm, CI, o00 + u * 16 + nI, ks * 32, half);
      accs[u] = wmma_bf16(a, b, accs[u]);
    }
  }
  int slot = SLOT_OF[j];
#pragma unroll
  for (int u = 0; u < 4; u++) {
    int o0 = o00 + u * 16;
    int kvsel = o0 / 384, rem = o0 % 384;
    int h = rem / 48, cbase = rem % 48;      // block-uniform
    int o = o0 + nI, c = cbase + nI;
    float bias = (j == 0) ? bkv[o] : 0.f;
#pragma unroll
    for (int r = 0; r < 8; r++) {
      int t = t0 + r + 8 * half;
      int bI = t >> 8, kk = t & 255;
      float v = accs[u][r] + bias;
      if (kvsel == 0) {
        if (slot >= 0)
          Kaug[(((size_t)(bI * HH + h)) * NK + kk) * DA + slot * 48 + c] = (bf16)v;
      } else {
        Vt[((size_t)(bI * HH + h) * 768 + (c * 16 + j)) * NK + kk] = (bf16)v;
      }
    }
  }
}

// ---------- per-key bias: -softplus(daa)*|kp|^2/C ----------
__global__ __launch_bounds__(256) void k_kbias(const bf16* __restrict__ Kaug,
                                               const float* __restrict__ sp,
                                               float* __restrict__ kbias) {
  int idx = blockIdx.x * 256 + threadIdx.x;  // [bh*256 + k]
  int bh = idx >> 8, h = bh & 7;
  const bf16* p = Kaug + (size_t)idx * DA + 8 * 48;
  float s = 0.f;
  for (int m = 0; m < 144; m++) { float v = (float)p[m]; s += v * v; }
  kbias[idx] = -sp[h] * s / 48.f;
}

// ---------- fused attention ----------
// Block = 2 waves; each wave owns 16 q rows. K-slices / V-tiles are staged in
// LDS with DOUBLE-BUFFERED async global->LDS copies: while computing slice n,
// slice n+1's transfers are in flight; s_wait_asynccnt(N) waits only for the
// older stage (each wave keeps exactly N newer async ops outstanding).
// The 32KB K double-buffer region is reused as the V double-buffer in phase 2.
// NOTE: buffer selection uses ternaries on loop parity (no arrays of
// LDS-derived pointers -> no addrspacecast static initializers for ld.lld).
__global__ __launch_bounds__(64) void k_attn(const bf16* __restrict__ Qaug,
                                             const bf16* __restrict__ Kaug,
                                             const float* __restrict__ kbias,
                                             const bf16* __restrict__ Vt,
                                             bf16* __restrict__ Obuf) {
  __shared__ __align__(32) bf16 pool[16384];          // 32 KB: K0|K1, later V0|V1
  __shared__ __align__(32) bf16 Plds[2][16][256];     // 16 KB: probabilities per wave
  __shared__ float invs[2][16];
  __shared__ float kb[256];

  int bh = blockIdx.x, bI = bh >> 3, h = bh & 7;
  int tid = threadIdx.x, w = tid >> 5, lane = tid & 31;
  int half = lane >> 4, nI = lane & 15;
  for (int i = tid; i < 256; i += 64) kb[i] = kbias[bh * NK + i];

  int q0 = blockIdx.y * 32 + w * 16;
  const bf16* Qb = Qaug + (size_t)bh * NQ * DA;
  const bf16* Kb = Kaug + (size_t)bh * NK * DA;
  const bf16* Vb = Vt + (size_t)bh * 768 * NK;

  // stage K-slice `ks` (256 keys x 32 d = 16KB; 16 async ops per wave)
  auto stage_k = [&](int ks, bf16* dst) {
    const bf16* src = Kb + ks * 32;
#pragma unroll
    for (int ch = 0; ch < 16; ch++) {
      int cidx = tid + ch * 64;
      int row = cidx >> 2, part = cidx & 3;
      async_copy16(dst + row * 32 + part * 8, src + (size_t)row * DA + part * 8);
    }
  };
  // stage V-tile `nt` (16 d2 rows x 256 keys = 8KB; 8 async ops per wave)
  auto stage_v = [&](int nt, bf16* dst) {
#pragma unroll
    for (int ch = 0; ch < 8; ch++) {
      int cidx = tid + ch * 64;
      int row = cidx >> 5, part = cidx & 31;
      async_copy16(dst + row * 256 + part * 8, Vb + (size_t)(nt * 16 + row) * 256 + part * 8);
    }
  };

  v8f accs[16];
#pragma unroll
  for (int kt = 0; kt < 16; kt++) accs[kt] = (v8f)V8F_ZERO;
  __syncthreads();

  // ---- GEMM1: logits = Qaug (16x544) * Kaug^T (544x256), double-buffered K ----
  stage_k(0, pool);
  for (int ks = 0; ks < 17; ks++) {
    bf16* cur = (ks & 1) ? (pool + 8192) : pool;
    if (ks < 16) {
      bf16* nxt = (ks & 1) ? pool : (pool + 8192);
      stage_k(ks + 1, nxt);
      async_wait16();
    } else {
      async_wait0();
    }
    __syncthreads();                                   // stage `ks` visible to both waves
    v16bf a = load_a(Qb, DA, q0 + nI, ks * 32, half);
#pragma unroll
    for (int kt = 0; kt < 16; kt++) {
      v16bf b = load_b(cur, 32, kt * 16 + nI, 0, half);
      accs[kt] = wmma_bf16(a, b, accs[kt]);
    }
    __syncthreads();                                   // reads done before buffer reuse
  }

  // ---- row softmax (unnormalized P; normalize at epilogue) ----
#pragma unroll
  for (int r = 0; r < 8; r++) {
    float m = -3.0e38f;
#pragma unroll
    for (int kt = 0; kt < 16; kt++) {
      float l = accs[kt][r] + kb[kt * 16 + nI];
      accs[kt][r] = l;
      m = fmaxf(m, l);
    }
    m = fmaxf(m, __shfl_xor(m, 1));
    m = fmaxf(m, __shfl_xor(m, 2));
    m = fmaxf(m, __shfl_xor(m, 4));
    m = fmaxf(m, __shfl_xor(m, 8));
    float s = 0.f;
#pragma unroll
    for (int kt = 0; kt < 16; kt++) {
      float pp = __expf(accs[kt][r] - m);
      s += pp;
      Plds[w][r + 8 * half][kt * 16 + nI] = (bf16)pp;
    }
    s += __shfl_xor(s, 1);
    s += __shfl_xor(s, 2);
    s += __shfl_xor(s, 4);
    s += __shfl_xor(s, 8);
    invs[w][r + 8 * half] = 1.f / s;
  }
  __syncthreads();   // K phase fully done; pool region now reused for V tiles

  // ---- GEMM2: O = P (16x256) * V (256x768), double-buffered V ----
  const bf16* Pb = &Plds[w][0][0];
  stage_v(0, pool);
  for (int nt = 0; nt < 48; nt++) {
    bf16* cur = (nt & 1) ? (pool + 4096) : pool;
    if (nt < 47) {
      bf16* nxt = (nt & 1) ? pool : (pool + 4096);
      stage_v(nt + 1, nxt);
      async_wait8();
    } else {
      async_wait0();
    }
    __syncthreads();
    v8f acc = V8F_ZERO;
#pragma unroll
    for (int ks = 0; ks < 8; ks++) {
      v16bf a = load_a(Pb, 256, nI, ks * 32, half);
      v16bf b = load_b(cur, 256, nI, ks * 32, half);
      acc = wmma_bf16(a, b, acc);
    }
    int n = nt * 16 + nI, c = n >> 4, j = n & 15;
#pragma unroll
    for (int r = 0; r < 8; r++) {
      int M = r + 8 * half, q = q0 + M;
      Obuf[(((size_t)(bI * NQ + q)) * 16 + j) * 384 + h * 48 + c] =
          (bf16)(acc[r] * invs[w][M]);
    }
    __syncthreads();
  }
}

// ---------- output projection + residual ----------
__global__ __launch_bounds__(32) void k_final(const bf16* __restrict__ Obuf,
                                              const bf16* __restrict__ wo,
                                              const float* __restrict__ bo,
                                              const float* __restrict__ hidden,
                                              float* __restrict__ out) {
  int t0 = blockIdx.x * 16, j = blockIdx.y;
  int g = GRADE[j];
  int lane = threadIdx.x & 31, half = lane >> 4, nI = lane & 15;
  const bf16* A = Obuf + j * 384;  // row stride 16*384
  const bf16* Bm = wo + (size_t)g * 48 * 384;
  v8f accs[3] = {V8F_ZERO, V8F_ZERO, V8F_ZERO};
#pragma unroll
  for (int ks = 0; ks < 12; ks++) {
    v16bf a = load_a(A, 16 * 384, t0 + nI, ks * 32, half);
#pragma unroll
    for (int u = 0; u < 3; u++) {
      v16bf b = load_b(Bm, 384, u * 16 + nI, ks * 32, half);
      accs[u] = wmma_bf16(a, b, accs[u]);
    }
  }
#pragma unroll
  for (int u = 0; u < 3; u++) {
    int o = u * 16 + nI;
    float bias = (j == 0) ? bo[o] : 0.f;
#pragma unroll
    for (int r = 0; r < 8; r++) {
      int t = t0 + r + 8 * half;
      size_t off = (size_t)t * 768 + o * 16 + j;
      out[off] = hidden[off] + accs[u][r] + bias;
    }
  }
}

extern "C" void kernel_launch(void* const* d_in, const int* in_sizes, int n_in,
                              void* d_out, int out_size, void* d_ws, size_t ws_size,
                              hipStream_t stream) {
  const float* hidden = (const float*)d_in[0];
  const float* vision = (const float*)d_in[1];
  const float* lnw    = (const float*)d_in[2];
  const float* Wq     = (const float*)d_in[3];
  const float* bq     = (const float*)d_in[4];
  const float* Wkv    = (const float*)d_in[5];
  const float* bkv    = (const float*)d_in[6];
  const float* Wo     = (const float*)d_in[7];
  const float* bo     = (const float*)d_in[8];
  const float* daa    = (const float*)d_in[9];
  float* out = (float*)d_out;

  char* p = (char*)d_ws;
  auto take = [&](size_t bytes) -> void* {
    void* r = (void*)p;
    p += (bytes + 255) & ~(size_t)255;
    return r;
  };
  bf16* normed = (bf16*)take((size_t)TQ * 16 * CI * 2);
  bf16* visbf  = (bf16*)take((size_t)TK * 16 * CI * 2);
  bf16* wqbf   = (bf16*)take((size_t)5 * 384 * CI * 2);
  bf16* wkvbf  = (bf16*)take((size_t)5 * 768 * CI * 2);
  bf16* wobf   = (bf16*)take((size_t)5 * 48 * 384 * 2);
  size_t qaugB = (size_t)BB * HH * NQ * DA * 2;
  size_t kaugB = (size_t)BB * HH * NK * DA * 2;
  bf16* Qaug   = (bf16*)take(qaugB);
  bf16* Kaug   = (bf16*)take(kaugB);
  float* kbias = (float*)take((size_t)BB * HH * NK * 4);
  float* spbuf = (float*)take((size_t)HH * 4);
  bf16* Vt     = (bf16*)take((size_t)BB * HH * 768 * NK * 2);
  bf16* Obuf   = (bf16*)take((size_t)TQ * 16 * 384 * 2);

  // zero pad columns of Qaug/Kaug (augmented GEMM requires zero padding)
  (void)hipMemsetAsync(Qaug, 0, qaugB, stream);
  (void)hipMemsetAsync(Kaug, 0, kaugB, stream);

  k_sp<<<1, 32, 0, stream>>>(daa, spbuf);
  k_packw<<<1800, 256, 0, stream>>>(Wq, Wkv, Wo, wqbf, wkvbf, wobf);
  k_norm<<<TQ, 256, 0, stream>>>(hidden, lnw, normed);
  k_packv<<<(TK * 16 * CI) / 256, 256, 0, stream>>>(vision, visbf);
  k_projq<<<dim3(TQ / 16, 384 / 64, 11), 32, 0, stream>>>(normed, wqbf, bq, spbuf, Qaug);
  k_projkv<<<dim3(TK / 16, 768 / 64, 16), 32, 0, stream>>>(visbf, wkvbf, bkv, Kaug, Vt);
  k_kbias<<<(BB * HH * NK) / 256, 256, 0, stream>>>(Kaug, spbuf, kbias);
  k_attn<<<dim3(BB * HH, NQ / 32), 64, 0, stream>>>(Qaug, Kaug, kbias, Vt, Obuf);
  k_final<<<dim3(TQ / 16, 16), 32, 0, stream>>>(Obuf, wobf, bo, hidden, out);
}
